// MagNetCrossAttentionPredictor_37709812859160
// MI455X (gfx1250) — matrix operand order repaired and checked by reading, not compiled, not run on hardware
//
#include <hip/hip_runtime.h>
#include <hip/hip_fp16.h>

typedef __attribute__((ext_vector_type(16))) _Float16 v16h;
typedef __attribute__((ext_vector_type(8)))  float    v8f;

#define NUM_HEADS 4
#define HIDDEN    128
#define EDIM      256
#define LTOK      8
#define NCARDS    10000
#define BATCH     16384

// ---- per-wave LDS slice layout (bytes) ----
#define OFF_EMB_MY 0        // 16 x 256 f16
#define OFF_EMB_OP 8192     // 16 x 256 f16
#define OFF_Q      16384    // 16 x 256 f16 (reused as attention output)
#define OFF_K      24576    // 16 x 256 f16
#define OFF_V      32768    // 16 x 256 f16
#define OFF_P      40960    // 2*4*8*8 f32 softmax probs
#define OFF_FEAT   43008    // 2 x 512 f32
#define OFF_H      47104    // 2 x 256 f32
#define OFF_H2     49152    // 2 x 64  f32
#define OFF_CARDS  49664    // 32 ints ([0..15]=my, [16..31]=op)
#define WAVE_LDS   50176    // 16B-aligned slice stride; 4 waves -> ~200KB/WGP (<320KB)

// ---------------- prep kernels ----------------

__global__ void card_feat_kernel(const float* __restrict__ xr,
                                 const float* __restrict__ xi,
                                 _Float16* __restrict__ cardF) {
  int i = blockIdx.x * blockDim.x + threadIdx.x;
  if (i >= NCARDS * HIDDEN) return;
  int c = i >> 7, d = i & 127;
  float r = xr[i], im = xi[i];
  cardF[c * EDIM + d]          = (_Float16)sqrtf(r * r + im * im + 1e-8f);
  cardF[c * EDIM + HIDDEN + d] = (_Float16)atan2f(im, r);
}

__global__ void f32_to_f16_kernel(const float* __restrict__ src,
                                  _Float16* __restrict__ dst, int n) {
  int i = blockIdx.x * blockDim.x + threadIdx.x;
  if (i < n) dst[i] = (_Float16)src[i];
}

// ---------------- WMMA tile loaders (ISA 7.12.2 layouts) ----------------
// All LDS access goes through extern-shared aliases indexed by integer
// offsets -> direct addrspace(3) GEPs -> guaranteed ds_* instructions.
// Helpers are __forceinline__ so weight pointers trace to kernargs ->
// global_* instructions (split LOADcnt / DScnt, no flat serialization).

// A: 16x32 f16 tile from row-major 16x256 LDS buffer (half-index aH).
// lanes 0-15: row M=lane, K = {k0+0..7, k0+16..23}; lanes 16-31: row M=lane-16, K+8.
__device__ __forceinline__ v16h load_a_tile(unsigned aH, int k0, int lane) {
  extern __shared__ _Float16 s_h[];
  unsigned p = aH + (unsigned)((lane & 15) * EDIM + k0 + ((lane >> 4) << 3));
  v16h a;
#pragma unroll
  for (int i = 0; i < 8; ++i) { a[i] = s_h[p + i]; a[i + 8] = s_h[p + i + 16]; }
  return a;
}

// B: 32x16 f16 tile where B[k][n] = W[n][k] (W row-major global, ld=256).
// lane n holds K = {k0+0..7, k0+16..23} of weight row n; lanes 16-31: K+8.
__device__ __forceinline__ v16h load_b_tile(const _Float16* __restrict__ wrow,
                                            int k0, int lane) {
  const _Float16* p = wrow + k0 + ((lane >> 4) << 3);
  v16h b;
#pragma unroll
  for (int i = 0; i < 8; ++i) { b[i] = p[i]; b[i + 8] = p[i + 16]; }
  return b;
}

// D(16x256) = A(16x256, LDS@aH) x W[wRowBase..+256)^T + bias ; store f16 LDS@dH.
__device__ __forceinline__ void wave_gemm_store(unsigned aH,
                                                const _Float16* __restrict__ W,
                                                const float* __restrict__ bias,
                                                int wRowBase, unsigned dH, int lane) {
  extern __shared__ _Float16 s_h[];
  int nlo = lane & 15;
  int mBase = (lane >> 4) << 3;
#pragma unroll 1
  for (int nt = 0; nt < 16; ++nt) {
    int wrow = wRowBase + nt * 16 + nlo;
    const _Float16* wp = W + wrow * EDIM;
    v8f acc = {};
#pragma unroll
    for (int kk = 0; kk < EDIM / 32; ++kk) {
      v16h a = load_a_tile(aH, kk * 32, lane);
      v16h b = load_b_tile(wp, kk * 32, lane);
      acc = __builtin_amdgcn_wmma_f32_16x16x32_f16(false, a, false, b,
                                                   (short)0, acc, false, false);
    }
    float bv = bias[wrow];
    unsigned col = dH + (unsigned)(nt * 16 + nlo);
#pragma unroll
    for (int v = 0; v < 8; ++v)
      s_h[col + (mBase + v) * EDIM] = (_Float16)(acc[v] + bv);
  }
}

// Out-projection + token mean: the f32 C/D layout gives each lane all 8
// token-rows of one batch element in its 8 acc VGPRs -> mean = register sum.
__device__ __forceinline__ void wave_gemm_mean(unsigned aH,
                                               const _Float16* __restrict__ W,
                                               const float* __restrict__ bias,
                                               unsigned featF, int featBase, int lane) {
  extern __shared__ float s_f[];
  int nlo = lane & 15;
  int e = lane >> 4;
#pragma unroll 1
  for (int nt = 0; nt < 16; ++nt) {
    int wrow = nt * 16 + nlo;
    const _Float16* wp = W + wrow * EDIM;
    v8f acc = {};
#pragma unroll
    for (int kk = 0; kk < EDIM / 32; ++kk) {
      v16h a = load_a_tile(aH, kk * 32, lane);
      v16h b = load_b_tile(wp, kk * 32, lane);
      acc = __builtin_amdgcn_wmma_f32_16x16x32_f16(false, a, false, b,
                                                   (short)0, acc, false, false);
    }
    float s = 0.f;
#pragma unroll
    for (int v = 0; v < 8; ++v) s += acc[v];
    s_f[featF + e * 512 + featBase + wrow] = s * 0.125f + bias[wrow];
  }
}

// ---------------- fused main kernel ----------------
// 128 threads = 4 waves; each wave owns 2 batch elements (16 query rows).

__global__ __launch_bounds__(128) void magnet_fused_kernel(
    const int* __restrict__ my_decks, const int* __restrict__ op_decks,
    const int* __restrict__ adj, const _Float16* __restrict__ cardF,
    const _Float16* __restrict__ wqkv_my, const float* __restrict__ in_b_my,
    const _Float16* __restrict__ wout_my, const float* __restrict__ out_b_my,
    const _Float16* __restrict__ wqkv_op, const float* __restrict__ in_b_op,
    const _Float16* __restrict__ wout_op, const float* __restrict__ out_b_op,
    const float* __restrict__ w1, const float* __restrict__ b1,
    const float* __restrict__ ln_g, const float* __restrict__ ln_b,
    const float* __restrict__ w2, const float* __restrict__ b2,
    const float* __restrict__ w3, const float* __restrict__ b3,
    float* __restrict__ out) {
  extern __shared__ _Float16 s_h[];   // all extern-shared aliases share the
  extern __shared__ float    s_f[];   // dynamic-LDS base address
  extern __shared__ int      s_i[];
  extern __shared__ uint4    s_v4[];

  const int lane = threadIdx.x & 31;
  const int wave = threadIdx.x >> 5;
  const unsigned wB = (unsigned)wave * WAVE_LDS;      // per-wave byte base
  const unsigned hEmbMy = (wB + OFF_EMB_MY) >> 1;     // half indices
  const unsigned hEmbOp = (wB + OFF_EMB_OP) >> 1;
  const unsigned hQ = (wB + OFF_Q) >> 1;
  const unsigned hK = (wB + OFF_K) >> 1;
  const unsigned hV = (wB + OFF_V) >> 1;
  const unsigned fP = (wB + OFF_P) >> 2;              // float indices
  const unsigned fFeat = (wB + OFF_FEAT) >> 2;
  const unsigned fH = (wB + OFF_H) >> 2;
  const unsigned fH2 = (wB + OFF_H2) >> 2;
  const unsigned iCards = (wB + OFF_CARDS) >> 2;      // int index

  const int b0 = (blockIdx.x * 4 + wave) * 2;

  if (lane < 16) {
    int e = lane >> 3, tok = lane & 7;
    s_i[iCards + lane]      = my_decks[(b0 + e) * LTOK + tok];
    s_i[iCards + 16 + lane] = op_decks[(b0 + e) * LTOK + tok];
  }
  // gather 16 embedding rows per side: 32 x 16B chunks per row
  const unsigned v4My = (wB + OFF_EMB_MY) >> 4;
  const unsigned v4Op = (wB + OFF_EMB_OP) >> 4;
  for (int t = lane; t < 512; t += 32) {
    int row = t >> 5, cc = t & 31;
    int e = row >> 3, tok = row & 7;
    int cm = my_decks[(b0 + e) * LTOK + tok];
    int co = op_decks[(b0 + e) * LTOK + tok];
    s_v4[v4My + row * 32 + cc] = ((const uint4*)(cardF + (size_t)cm * EDIM))[cc];
    s_v4[v4Op + row * 32 + cc] = ((const uint4*)(cardF + (size_t)co * EDIM))[cc];
  }
  __syncthreads();

#pragma unroll 1
  for (int pass = 0; pass < 2; ++pass) {
    const unsigned  hQE = pass ? hEmbOp : hEmbMy;
    const unsigned  hKV = pass ? hEmbMy : hEmbOp;
    const _Float16* Wqkv = pass ? wqkv_op : wqkv_my;
    const float*    inB  = pass ? in_b_op : in_b_my;
    const _Float16* Wout = pass ? wout_op : wout_my;
    const float*    outB = pass ? out_b_op : out_b_my;

    wave_gemm_store(hQE, Wqkv, inB,   0, hQ, lane);  // Q (rows 0..255 of in_w)
    wave_gemm_store(hKV, Wqkv, inB, 256, hK, lane);  // K
    wave_gemm_store(hKV, Wqkv, inB, 512, hV, lane);  // V
    __syncthreads();

    // scores + adjacency bias + softmax : 64 (elem,head,row) items, 2 per lane
#pragma unroll 1
    for (int t = 0; t < 2; ++t) {
      int item = lane + 32 * t;
      int e = item >> 5, h = (item >> 3) & 3, q = item & 7;
      unsigned qrow = hQ + (unsigned)((e * 8 + q) * EDIM + h * 64);
      float s[8], mx = -3.4e38f;
#pragma unroll 1
      for (int k = 0; k < 8; ++k) {
        unsigned krow = hK + (unsigned)((e * 8 + k) * EDIM + h * 64);
        float acc = 0.f;
#pragma unroll
        for (int d = 0; d < 64; ++d)
          acc += (float)s_h[qrow + d] * (float)s_h[krow + d];
        long ai = (pass == 0)
            ? (long)s_i[iCards + e * 8 + q] * NCARDS + s_i[iCards + 16 + e * 8 + k]
            : (long)s_i[iCards + e * 8 + k] * NCARDS + s_i[iCards + 16 + e * 8 + q];
        float bias = (adj[ai] > 0) ? 0.f : -10000.f;
        s[k] = acc * 0.125f + bias;
        mx = fmaxf(mx, s[k]);
      }
      float sum = 0.f;
#pragma unroll
      for (int k = 0; k < 8; ++k) { s[k] = __expf(s[k] - mx); sum += s[k]; }
      float inv = 1.f / sum;
#pragma unroll
      for (int k = 0; k < 8; ++k) s_f[fP + item * 8 + k] = s[k] * inv;
    }
    __syncthreads();

    // P @ V  -> overwrite Q buffer with attention output (f16)
#pragma unroll 1
    for (int t = 0; t < 2; ++t) {
      int item = lane + 32 * t;
      int e = item >> 5, h = (item >> 3) & 3, q = item & 7;
      float p[8];
#pragma unroll
      for (int k = 0; k < 8; ++k) p[k] = s_f[fP + item * 8 + k];
      unsigned orow = hQ + (unsigned)((e * 8 + q) * EDIM + h * 64);
      unsigned vrow = hV + (unsigned)(e * 8 * EDIM + h * 64);
#pragma unroll 1
      for (int d = 0; d < 64; ++d) {
        float acc = 0.f;
#pragma unroll
        for (int k = 0; k < 8; ++k)
          acc += p[k] * (float)s_h[vrow + k * EDIM + d];
        s_h[orow + d] = (_Float16)acc;
      }
    }
    __syncthreads();

    wave_gemm_mean(hQ, Wout, outB, fFeat, pass * 256, lane);
    __syncthreads();
  }

  // ---------------- MLP head (2 elements per wave) ----------------
#pragma unroll 1
  for (int e = 0; e < 2; ++e) {
    for (int j = lane; j < 256; j += 32) {
      const float* wr = w1 + j * 512;
      float acc = b1[j];
#pragma unroll 4
      for (int k = 0; k < 512; ++k) acc += s_f[fFeat + e * 512 + k] * wr[k];
      s_f[fH + e * 256 + j] = acc;
    }
  }
  __syncthreads();

#pragma unroll 1
  for (int e = 0; e < 2; ++e) {
    float ps = 0.f, ps2 = 0.f;
    for (int j = lane; j < 256; j += 32) {
      float v = s_f[fH + e * 256 + j];
      ps += v; ps2 += v * v;
    }
#pragma unroll
    for (int off = 16; off > 0; off >>= 1) {
      ps  += __shfl_xor(ps,  off, 32);
      ps2 += __shfl_xor(ps2, off, 32);
    }
    float mu  = ps * (1.f / 256.f);
    float var = ps2 * (1.f / 256.f) - mu * mu;
    float inv = rsqrtf(var + 1e-5f);
    for (int j = lane; j < 256; j += 32) {
      float v = (s_f[fH + e * 256 + j] - mu) * inv * ln_g[j] + ln_b[j];
      s_f[fH + e * 256 + j] = fmaxf(v, 0.f);
    }
    __syncthreads();
    for (int j = lane; j < 64; j += 32) {
      const float* wr = w2 + j * 256;
      float acc = b2[j];
#pragma unroll 4
      for (int k = 0; k < 256; ++k) acc += s_f[fH + e * 256 + k] * wr[k];
      s_f[fH2 + e * 64 + j] = fmaxf(acc, 0.f);
    }
    __syncthreads();
    float pl = 0.f;
    for (int k = lane; k < 64; k += 32) pl += s_f[fH2 + e * 64 + k] * w3[k];
#pragma unroll
    for (int off = 16; off > 0; off >>= 1) pl += __shfl_xor(pl, off, 32);
    if (lane == 0) out[b0 + e] = pl + b3[0];
  }
}

// ---------------- launch ----------------

extern "C" void kernel_launch(void* const* d_in, const int* in_sizes, int n_in,
                              void* d_out, int out_size, void* d_ws, size_t ws_size,
                              hipStream_t stream) {
  (void)in_sizes; (void)n_in; (void)out_size; (void)ws_size;
  const float* x_real   = (const float*)d_in[0];
  const float* x_imag   = (const float*)d_in[1];
  const int*   my_decks = (const int*)d_in[2];
  const int*   op_decks = (const int*)d_in[3];
  const int*   adj      = (const int*)d_in[4];
  const float* in_w_my  = (const float*)d_in[5];
  const float* in_b_my  = (const float*)d_in[6];
  const float* out_w_my = (const float*)d_in[7];
  const float* out_b_my = (const float*)d_in[8];
  const float* in_w_op  = (const float*)d_in[9];
  const float* in_b_op  = (const float*)d_in[10];
  const float* out_w_op = (const float*)d_in[11];
  const float* out_b_op = (const float*)d_in[12];
  const float* w1 = (const float*)d_in[13];
  const float* b1 = (const float*)d_in[14];
  const float* ln_g = (const float*)d_in[15];
  const float* ln_b = (const float*)d_in[16];
  const float* w2 = (const float*)d_in[17];
  const float* b2 = (const float*)d_in[18];
  const float* w3 = (const float*)d_in[19];
  const float* b3 = (const float*)d_in[20];
  float* out = (float*)d_out;

  char* ws = (char*)d_ws;
  _Float16* cardF = (_Float16*)ws;
  size_t off = (size_t)NCARDS * EDIM * sizeof(_Float16);
  _Float16* wqkv_my_h = (_Float16*)(ws + off); off += (size_t)768 * 256 * 2;
  _Float16* wqkv_op_h = (_Float16*)(ws + off); off += (size_t)768 * 256 * 2;
  _Float16* wout_my_h = (_Float16*)(ws + off); off += (size_t)256 * 256 * 2;
  _Float16* wout_op_h = (_Float16*)(ws + off); // total ~6MB of workspace

  card_feat_kernel<<<(NCARDS * HIDDEN + 255) / 256, 256, 0, stream>>>(x_real, x_imag, cardF);
  f32_to_f16_kernel<<<(768 * 256 + 255) / 256, 256, 0, stream>>>(in_w_my, wqkv_my_h, 768 * 256);
  f32_to_f16_kernel<<<(768 * 256 + 255) / 256, 256, 0, stream>>>(in_w_op, wqkv_op_h, 768 * 256);
  f32_to_f16_kernel<<<(256 * 256 + 255) / 256, 256, 0, stream>>>(out_w_my, wout_my_h, 256 * 256);
  f32_to_f16_kernel<<<(256 * 256 + 255) / 256, 256, 0, stream>>>(out_w_op, wout_op_h, 256 * 256);

  hipFuncSetAttribute((const void*)magnet_fused_kernel,
                      hipFuncAttributeMaxDynamicSharedMemorySize, 4 * WAVE_LDS);

  dim3 grid(BATCH / 8), block(128);
  magnet_fused_kernel<<<grid, block, 4 * WAVE_LDS, stream>>>(
      my_decks, op_decks, adj, cardF,
      wqkv_my_h, in_b_my, wout_my_h, out_b_my,
      wqkv_op_h, in_b_op, wout_op_h, out_b_op,
      w1, b1, ln_g, ln_b, w2, b2, w3, b3, out);
}
